// SwinTransformerBlock_42296837931593
// MI455X (gfx1250) — compile-verified, matrix operand order checked by reading, not run on hardware
//
#include <hip/hip_runtime.h>
#include <hip/hip_bf16.h>

// ---------------------------------------------------------------------------
// Swin Transformer block on gfx1250 (MI455X).
// WMMA f16 (f32 accum) for all matmuls; double-buffered async-to-LDS / TDM
// staging pipelines (guarded, with manual fallback).
// B=4, H=W=56, C=128, NH=4, DK=32, WS=7, SHIFT=3.
// ---------------------------------------------------------------------------

typedef _Float16 h8   __attribute__((ext_vector_type(8)));
typedef _Float16 v16h __attribute__((ext_vector_type(16)));
typedef float    v8f  __attribute__((ext_vector_type(8)));

#define F_F16  1
#define F_GELU 2
#define F_RES  4
#define F_SCAT 8

// ---- gfx1250 async / TDM plumbing (guarded) -------------------------------
typedef int v4i __attribute__((vector_size(16)));
typedef __attribute__((address_space(1))) v4i* gv4i_p;   // global b128 chunk
typedef __attribute__((address_space(3))) v4i* lv4i_p;   // LDS b128 chunk
#define GP128(p) ((gv4i_p)(const_cast<void*>((const void*)(p))))
#define LP128(p) ((lv4i_p)(void*)(p))
#define AS3V(p)  ((__attribute__((address_space(3))) void*)(p))
#define LDSOFF(p) ((unsigned)(unsigned long long)AS3V(p))

#if __has_builtin(__builtin_amdgcn_global_load_async_to_lds_b128)
#define HAVE_ASYNC 1
#endif
#if __has_builtin(__builtin_amdgcn_s_wait_asynccnt)
#define WAIT_ASYNC() __builtin_amdgcn_s_wait_asynccnt(0)
#else
#define WAIT_ASYNC() asm volatile("s_wait_asynccnt 0x0" ::: "memory")
#endif

#if __has_builtin(__builtin_amdgcn_tensor_load_to_lds)
#define HAVE_TDM 1
typedef unsigned int u32x4 __attribute__((vector_size(16)));
typedef int          i32x4 __attribute__((vector_size(16)));
typedef int          i32x8 __attribute__((vector_size(32)));

// 2D f16 tile load via Tensor Data Mover. tile_w/tensor_w in elements.
static __device__ __forceinline__ void tdm_load_tile_f16(
        unsigned lds_off, const _Float16* gsrc,
        unsigned tile_w, unsigned tile_h,
        unsigned tensor_w, unsigned tensor_h, unsigned row_stride) {
    unsigned long long ga = (unsigned long long)(size_t)gsrc;
    u32x4 g0;
    g0[0] = 1u;                                            // count=1 (valid user D#)
    g0[1] = lds_off;                                       // lds_addr (bytes)
    g0[2] = (unsigned)ga;                                  // global_addr[31:0]
    g0[3] = (unsigned)((ga >> 32) & 0x01FFFFFFu) | (2u << 30);  // addr[56:32] | type=2
    i32x8 g1;
    g1[0] = 1 << 16;                                       // data_size=1 (2B), wg_mask=0
    g1[1] = (int)((tensor_w & 0xFFFFu) << 16);             // tensor_dim0[15:0]
    g1[2] = (int)((tensor_w >> 16) | ((tensor_h & 0xFFFFu) << 16)); // dim0 hi | dim1 lo
    g1[3] = (int)((tensor_h >> 16) | (tile_w << 16));      // dim1 hi | tile_dim0
    g1[4] = (int)tile_h;                                   // tile_dim1 (tile_dim2=0)
    g1[5] = (int)row_stride;                               // tensor_dim0_stride[31:0]
    g1[6] = 0;
    g1[7] = 0;
    i32x4 gz4 = {0, 0, 0, 0};
    i32x8 gz8 = {0, 0, 0, 0, 0, 0, 0, 0};
    __builtin_amdgcn_tensor_load_to_lds(g0, g1, gz4, gz4, gz8, 0);
}
#endif

static __device__ __forceinline__ v16h load_frag(const _Float16* p0, const _Float16* p1) {
    h8 r0 = *(const h8*)p0;
    h8 r1 = *(const h8*)p1;
    v16h a;
#pragma unroll
    for (int i = 0; i < 8; ++i) { a[i] = r0[i]; a[i + 8] = r1[i]; }
    return a;
}

static __device__ __forceinline__ v8f wmma_f16(v16h a, v16h b, v8f c) {
    return __builtin_amdgcn_wmma_f32_16x16x32_f16(false, a, false, b, (short)0, c,
                                                  false, false);
}

// ---------------------------------------------------------------------------
// Weight convert+transpose: W[K][N] f32 -> Wt[N][K] f16.
// ---------------------------------------------------------------------------
__global__ __launch_bounds__(256) void wtrans_kernel(const float* __restrict__ W,
                                                     _Float16* __restrict__ Wt,
                                                     int K, int N) {
    int idx = blockIdx.x * 256 + threadIdx.x;
    if (idx < N * K) {
        int n = idx / K, k = idx - n * K;
        Wt[idx] = (_Float16)W[(size_t)k * N + n];
    }
}

// ---------------------------------------------------------------------------
// Shift + window partition + f32->f16, pad 49 tokens -> 64 rows (zeros).
// xw: [256 windows][64][128] f16
// ---------------------------------------------------------------------------
__global__ __launch_bounds__(256) void shift_gather_kernel(const float* __restrict__ x,
                                                           _Float16* __restrict__ xw) {
    int w = blockIdx.x;              // 0..255
    int b = w >> 6, wi = w & 63, wr = wi >> 3, wc = wi & 7;
    for (int s = threadIdx.x; s < 64 * 128; s += 256) {
        int t = s >> 7, ch = s & 127;
        _Float16 v = (_Float16)0.f;
        if (t < 49) {
            int is = wr * 7 + t / 7, js = wc * 7 + t % 7;
            int i = (is + 3) % 56, j = (js + 3) % 56;
            v = (_Float16)x[(((size_t)(b * 56 + i)) * 56 + j) * 128 + ch];
        }
        xw[((size_t)w * 64 + t) * 128 + ch] = v;
    }
}

// ---------------------------------------------------------------------------
// Generic WMMA GEMM: Out[M,N] = A[M,K](f16) * Bt[N][K](f16, pre-transposed) + bias
// Block tile 128x64, 8 waves, each wave 32x32 (2x2 WMMA tiles), K step 32.
// Double-buffered LDS pipeline: async copy of tile i+1 overlaps WMMA of tile i.
// ---------------------------------------------------------------------------
#define GLDA 40
#define GLDB 40
__global__ __launch_bounds__(256) void gemm_kernel(const _Float16* __restrict__ A,
                                                   const _Float16* __restrict__ Bt,
                                                   const float* __restrict__ bias,
                                                   const float* __restrict__ resid,
                                                   void* __restrict__ Out,
                                                   int M, int N, int K, int flags) {
    __shared__ __align__(16) _Float16 a_s[2][128 * GLDA];
    __shared__ __align__(16) _Float16 b_s[2][64 * GLDB];

    const int tid = threadIdx.x;
    const int bm = blockIdx.y * 128;
    const int bn = blockIdx.x * 64;
    const int wv = tid >> 5, lane = tid & 31;
    const int wm = (wv >> 1) * 32, wn = (wv & 1) * 32;
    const int lm = lane & 15, lh = lane >> 4;

    v8f c[2][2] = {};

    auto stage = [&](int p, int kc) {
#ifdef HAVE_ASYNC
#pragma unroll
        for (int s = 0; s < 2; ++s) {
            int slot = tid + s * 256;
            int row = slot >> 2, seg = (slot & 3) * 8;
            __builtin_amdgcn_global_load_async_to_lds_b128(
                GP128(A + (size_t)(bm + row) * K + kc + seg),
                LP128(a_s[p] + row * GLDA + seg), 0, 0);
        }
        int n = tid >> 2, seg = (tid & 3) * 8;
        __builtin_amdgcn_global_load_async_to_lds_b128(
            GP128(Bt + (size_t)(bn + n) * K + kc + seg),
            LP128(b_s[p] + n * GLDB + seg), 0, 0);
#else
#pragma unroll
        for (int s = 0; s < 2; ++s) {
            int slot = tid + s * 256;
            int row = slot >> 2, seg = (slot & 3) * 8;
            *(h8*)(a_s[p] + row * GLDA + seg) =
                *(const h8*)(A + (size_t)(bm + row) * K + kc + seg);
        }
        int n = tid >> 2, seg = (tid & 3) * 8;
        *(h8*)(b_s[p] + n * GLDB + seg) =
            *(const h8*)(Bt + (size_t)(bn + n) * K + kc + seg);
#endif
    };

    stage(0, 0);
#ifdef HAVE_ASYNC
    WAIT_ASYNC();
#endif
    __syncthreads();

    int p = 0;
    for (int kc = 0; kc < K; kc += 32) {
        if (kc + 32 < K) stage(p ^ 1, kc + 32);    // overlap copy of next tile
        if (kc + 64 < K) {                         // prefetch two tiles ahead
            int row = tid >> 2, seg = (tid & 3) * 8;
            __builtin_prefetch(A + (size_t)(bm + row) * K + kc + 64 + seg, 0, 1);
        }

        v16h af[2], bf[2];
#pragma unroll
        for (int i = 0; i < 2; ++i) {
            const _Float16* pa = a_s[p] + (wm + i * 16 + lm) * GLDA;
            af[i] = load_frag(pa + lh * 8, pa + 16 + lh * 8);
            const _Float16* pb = b_s[p] + (wn + i * 16 + lm) * GLDB;
            bf[i] = load_frag(pb + lh * 8, pb + 16 + lh * 8);
        }
#pragma unroll
        for (int i = 0; i < 2; ++i)
#pragma unroll
            for (int j = 0; j < 2; ++j)
                c[i][j] = wmma_f16(af[i], bf[j], c[i][j]);

#ifdef HAVE_ASYNC
        WAIT_ASYNC();
#endif
        __syncthreads();
        p ^= 1;
    }

    // Epilogue.
#pragma unroll
    for (int i = 0; i < 2; ++i) {
#pragma unroll
        for (int j = 0; j < 2; ++j) {
#pragma unroll
            for (int r = 0; r < 8; ++r) {
                int gm = bm + wm + i * 16 + r + 8 * lh;
                int gn = bn + wn + j * 16 + lm;
                float v = c[i][j][r] + bias[gn];
                if (flags & F_GELU)
                    v = 0.5f * v * (1.f + erff(v * 0.70710678118654752f));
                if (flags & F_RES)
                    v += resid[(size_t)gm * 128 + gn];
                if (flags & F_SCAT) {
                    // window reverse + un-roll: gm = w*64 + t
                    int w = gm >> 6, t = gm & 63;
                    if (t < 49) {
                        int b = w >> 6, wi = w & 63, wr = wi >> 3, wc = wi & 7;
                        int is = wr * 7 + t / 7, js = wc * 7 + t % 7;
                        int ii = (is + 3) % 56, jj = (js + 3) % 56;
                        ((float*)Out)[(((size_t)(b * 56 + ii)) * 56 + jj) * 128 + gn] = v;
                    }
                } else if (flags & F_F16) {
                    ((_Float16*)Out)[(size_t)gm * N + gn] = (_Float16)v;
                } else {
                    ((float*)Out)[(size_t)gm * N + gn] = v;
                }
            }
        }
    }
}

// ---------------------------------------------------------------------------
// Window attention: one workgroup per (window, head-pair).
// qw/kw/vw: [256][64][128] f16 (padded rows 49..63). aw: same layout (output).
// ---------------------------------------------------------------------------
__global__ __launch_bounds__(256) void win_attn_kernel(const _Float16* __restrict__ qw,
                                                       const _Float16* __restrict__ kw,
                                                       const _Float16* __restrict__ vw,
                                                       _Float16* __restrict__ aw) {
    __shared__ float    s_s[2 * 64 * 66];
    __shared__ __align__(16) _Float16 p_s[2 * 64 * 64];
    __shared__ __align__(16) _Float16 vt_s[2 * 32 * 64];

    const int w = blockIdx.x >> 1;
    const int hp = blockIdx.x & 1;                // heads hp*2, hp*2+1
    const int tid = threadIdx.x, wv = tid >> 5, lane = tid & 31;
    const int lm = lane & 15, lh = lane >> 4;
    const float scale = 0.17677669529663687f;     // 1/sqrt(32)

    // Stage V^T per head: vt_s[hh][d][key]
    for (int s2 = tid; s2 < 2 * 32 * 64; s2 += 256) {
        int hh = s2 >> 11, d = (s2 >> 6) & 31, k = s2 & 63;
        vt_s[(hh * 32 + d) * 64 + k] =
            vw[((size_t)w * 64 + k) * 128 + (hp * 2 + hh) * 32 + d];
    }

    // Scores: 2 heads x 16 tiles = 32; 4 per wave. K-dim = 32 -> one WMMA.
#pragma unroll
    for (int u = 0; u < 4; ++u) {
        int tt = wv * 4 + u;
        int hh = tt >> 4, rem = tt & 15, mt = rem >> 2, nt = rem & 3;
        int head = hp * 2 + hh;
        const _Float16* pq = qw + ((size_t)w * 64 + mt * 16 + lm) * 128 + head * 32;
        v16h a = load_frag(pq + lh * 8, pq + 16 + lh * 8);
        const _Float16* pk = kw + ((size_t)w * 64 + nt * 16 + lm) * 128 + head * 32;
        v16h b = load_frag(pk + lh * 8, pk + 16 + lh * 8);
        v8f c = {};
        c = wmma_f16(a, b, c);
#pragma unroll
        for (int r = 0; r < 8; ++r)
            s_s[(hh * 64 + mt * 16 + r + 8 * lh) * 66 + nt * 16 + lm] = c[r] * scale;
    }
    __syncthreads();

    // Softmax over keys < 49, normalized P stored f16; pad keys -> 0.
    if (tid < 128) {
        int hh = tid >> 6, row = tid & 63;
        float* sr = s_s + (hh * 64 + row) * 66;
        float mx = -1e30f;
        for (int j = 0; j < 49; ++j) mx = fmaxf(mx, sr[j]);
        float sum = 0.f;
        for (int j = 0; j < 49; ++j) { float e = __expf(sr[j] - mx); sr[j] = e; sum += e; }
        float inv = 1.f / sum;
        _Float16* pr = p_s + (hh * 64 + row) * 64;
        for (int j = 0; j < 49; ++j) pr[j] = (_Float16)(sr[j] * inv);
        for (int j = 49; j < 64; ++j) pr[j] = (_Float16)0.f;
    }
    __syncthreads();

    // Out = P @ V : 2 heads x (4 mt x 2 nt) = 16 tiles; 2 per wave; K=64 -> 2 WMMA.
#pragma unroll
    for (int u = 0; u < 2; ++u) {
        int tt = wv * 2 + u;
        int hh = tt >> 3, rem = tt & 7, mt = rem >> 1, nt = rem & 1;
        int head = hp * 2 + hh;
        v8f c = {};
#pragma unroll
        for (int kc = 0; kc < 64; kc += 32) {
            const _Float16* pp = p_s + (hh * 64 + mt * 16 + lm) * 64 + kc;
            v16h a = load_frag(pp + lh * 8, pp + 16 + lh * 8);
            const _Float16* pv = vt_s + (hh * 32 + nt * 16 + lm) * 64 + kc;
            v16h b = load_frag(pv + lh * 8, pv + 16 + lh * 8);
            c = wmma_f16(a, b, c);
        }
#pragma unroll
        for (int r = 0; r < 8; ++r)
            aw[((size_t)w * 64 + mt * 16 + r + 8 * lh) * 128 + head * 32 + nt * 16 + lm] =
                (_Float16)c[r];
    }
}

// ---------------------------------------------------------------------------
// LayerNorm (per 128-elem row), f32 in -> f16 out. One wave per row.
// ---------------------------------------------------------------------------
__global__ __launch_bounds__(256) void layernorm_kernel(const float* __restrict__ x,
                                                        const float* __restrict__ g,
                                                        const float* __restrict__ be,
                                                        _Float16* __restrict__ out) {
    int row = blockIdx.x * 8 + (threadIdx.x >> 5);
    int lane = threadIdx.x & 31;
    const float* xr = x + (size_t)row * 128;
    float v[4], s = 0.f;
#pragma unroll
    for (int i = 0; i < 4; ++i) { v[i] = xr[lane + 32 * i]; s += v[i]; }
#pragma unroll
    for (int off = 16; off; off >>= 1) s += __shfl_xor(s, off, 32);
    float mean = s * (1.f / 128.f);
    float q = 0.f;
#pragma unroll
    for (int i = 0; i < 4; ++i) { float d = v[i] - mean; q += d * d; }
#pragma unroll
    for (int off = 16; off; off >>= 1) q += __shfl_xor(q, off, 32);
    float inv = rsqrtf(q * (1.f / 128.f) + 1e-5f);
#pragma unroll
    for (int i = 0; i < 4; ++i) {
        int cCol = lane + 32 * i;
        out[(size_t)row * 128 + cCol] = (_Float16)((v[i] - mean) * inv * g[cCol] + be[cCol]);
    }
}

// ---------------------------------------------------------------------------
// Global attention (flash-style, online softmax).
// One workgroup per (batch, head, 64-query block). Seq = 3136 = 49 * 64.
// q2/k2/v2/a2: [12544][128] f16 (token = b*3136 + s; head slice 32 cols).
// K tiles staged via TDM; K/V double-buffered; softmax 4 threads/row.
// ---------------------------------------------------------------------------
__global__ __launch_bounds__(256) void global_attn_kernel(const _Float16* __restrict__ q2,
                                                          const _Float16* __restrict__ k2,
                                                          const _Float16* __restrict__ v2,
                                                          _Float16* __restrict__ a2) {
    __shared__ __align__(16) _Float16 qs[64 * 32], ks[2][64 * 32], vt[2][32 * 64], ps[64 * 64];
    __shared__ float ss[64 * 66], os[64 * 32], ms[64], ls[64];

    const int bid = blockIdx.x;
    const int qb = bid % 49;
    const int h = (bid / 49) & 3;
    const int b = bid / 196;
    const int tid = threadIdx.x, wv = tid >> 5, lane = tid & 31;
    const int lm = lane & 15, lh = lane >> 4;
    const size_t base = (size_t)b * 3136;
    const float scale = 0.17677669529663687f;

    auto stage_k = [&](int p, int kb) {
#ifdef HAVE_TDM
        if (tid == 0)
            tdm_load_tile_f16(LDSOFF(ks[p]), k2 + (base + kb * 64) * 128 + h * 32,
                              32, 64, 32, 64, 128);
#else
        int row = tid >> 2, seg = (tid & 3) * 8;
        *(h8*)(ks[p] + row * 32 + seg) =
            *(const h8*)(k2 + (base + kb * 64 + row) * 128 + h * 32 + seg);
#endif
        // V chunk transposed: vt[p][d][key]
        for (int s2 = tid; s2 < 32 * 64; s2 += 256) {
            int d = s2 >> 6, k = s2 & 63;
            vt[p][d * 64 + k] = v2[(base + kb * 64 + k) * 128 + h * 32 + d];
        }
    };

#ifdef HAVE_TDM
    if (tid == 0)
        tdm_load_tile_f16(LDSOFF(qs), q2 + (base + qb * 64) * 128 + h * 32,
                          /*tile*/32, 64, /*tensor*/32, 64, /*stride*/128);
#else
    {   // load Q block once: 64x32 halfs, 4 threads/row (16B chunks)
        int row = tid >> 2, seg = (tid & 3) * 8;
        *(h8*)(qs + row * 32 + seg) =
            *(const h8*)(q2 + (base + qb * 64 + row) * 128 + h * 32 + seg);
    }
#endif
    stage_k(0, 0);
    if (tid < 64) { ms[tid] = -1e30f; ls[tid] = 0.f; }
    for (int i = tid; i < 64 * 32; i += 256) os[i] = 0.f;
#ifdef HAVE_TDM
    __builtin_amdgcn_s_wait_tensorcnt(0);
#endif
    __syncthreads();

    int p = 0;
    for (int kb = 0; kb < 49; ++kb) {
        if (kb + 1 < 49) stage_k(p ^ 1, kb + 1);   // overlap next K/V chunk

        // S = scale * Q K^T : 16 tiles, 2 per wave, one WMMA each (K=32).
#pragma unroll
        for (int u = 0; u < 2; ++u) {
            int tt = wv * 2 + u;
            int mt = tt >> 2, nt = tt & 3;
            const _Float16* pq = qs + (mt * 16 + lm) * 32;
            v16h a = load_frag(pq + lh * 8, pq + 16 + lh * 8);
            const _Float16* pk = ks[p] + (nt * 16 + lm) * 32;
            v16h bb = load_frag(pk + lh * 8, pk + 16 + lh * 8);
            v8f c = {};
            c = wmma_f16(a, bb, c);
#pragma unroll
            for (int r = 0; r < 8; ++r)
                ss[(mt * 16 + r + 8 * lh) * 66 + nt * 16 + lm] = c[r] * scale;
        }
        __syncthreads();

        // Online softmax update: 4 threads per row (16 cols each).
        {
            int row = tid >> 2, part = tid & 3;
            float* sr = ss + row * 66 + part * 16;
            float mx = -1e30f;
#pragma unroll
            for (int j = 0; j < 16; ++j) mx = fmaxf(mx, sr[j]);
            mx = fmaxf(mx, __shfl_xor(mx, 1, 32));
            mx = fmaxf(mx, __shfl_xor(mx, 2, 32));
            float mold = ms[row];
            float mnew = fmaxf(mold, mx);
            float alpha = __expf(mold - mnew);
            float sum = 0.f;
            _Float16* pr = ps + row * 64 + part * 16;
#pragma unroll
            for (int j = 0; j < 16; ++j) {
                float e = __expf(sr[j] - mnew);
                pr[j] = (_Float16)e;
                sum += e;
            }
            sum += __shfl_xor(sum, 1, 32);
            sum += __shfl_xor(sum, 2, 32);
            if (part == 0) { ls[row] = ls[row] * alpha + sum; ms[row] = mnew; }
            float* orow = os + row * 32 + part * 8;
#pragma unroll
            for (int j = 0; j < 8; ++j) orow[j] *= alpha;
        }
        __syncthreads();

        // O += P @ V : 8 tiles (4x2), 1 per wave, K=64 -> 2 WMMA with C from LDS.
        {
            int mt = wv >> 1, nt = wv & 1;
            v8f c;
#pragma unroll
            for (int r = 0; r < 8; ++r)
                c[r] = os[(mt * 16 + r + 8 * lh) * 32 + nt * 16 + lm];
#pragma unroll
            for (int kc = 0; kc < 64; kc += 32) {
                const _Float16* pp = ps + (mt * 16 + lm) * 64 + kc;
                v16h a = load_frag(pp + lh * 8, pp + 16 + lh * 8);
                const _Float16* pv = vt[p] + (nt * 16 + lm) * 64 + kc;
                v16h bb = load_frag(pv + lh * 8, pv + 16 + lh * 8);
                c = wmma_f16(a, bb, c);
            }
#pragma unroll
            for (int r = 0; r < 8; ++r)
                os[(mt * 16 + r + 8 * lh) * 32 + nt * 16 + lm] = c[r];
        }
#ifdef HAVE_TDM
        __builtin_amdgcn_s_wait_tensorcnt(0);      // next K tile landed
#endif
        __syncthreads();
        p ^= 1;
    }

    {   // epilogue: 4 threads per row, 8 cols each
        int row = tid >> 2, part = tid & 3;
        float inv = 1.f / ls[row];
        _Float16* pout = a2 + (base + qb * 64 + row) * 128 + h * 32 + part * 8;
        float* orow = os + row * 32 + part * 8;
#pragma unroll
        for (int j = 0; j < 8; ++j) pout[j] = (_Float16)(orow[j] * inv);
    }
}

// ---------------------------------------------------------------------------
// Host-side orchestration.
// ---------------------------------------------------------------------------
extern "C" void kernel_launch(void* const* d_in, const int* in_sizes, int n_in,
                              void* d_out, int out_size, void* d_ws, size_t ws_size,
                              hipStream_t stream) {
    (void)in_sizes; (void)n_in; (void)out_size; (void)ws_size;
    const float* x  = (const float*)d_in[0];
    const float* wq = (const float*)d_in[1];  const float* bq = (const float*)d_in[2];
    const float* wk = (const float*)d_in[3];  const float* bk = (const float*)d_in[4];
    const float* wv = (const float*)d_in[5];  const float* bv = (const float*)d_in[6];
    const float* wo = (const float*)d_in[7];  const float* bo = (const float*)d_in[8];
    const float* w1 = (const float*)d_in[9];  const float* b1 = (const float*)d_in[10];
    const float* w2 = (const float*)d_in[11]; const float* b2 = (const float*)d_in[12];
    const float* g1 = (const float*)d_in[13]; const float* be1 = (const float*)d_in[14];
    const float* g2 = (const float*)d_in[15]; const float* be2 = (const float*)d_in[16];

    char* ws = (char*)d_ws;
    size_t off = 0;
    auto alloc = [&](size_t bytes) { size_t o = off; off += (bytes + 255) & ~(size_t)255; return o; };
    const size_t WIN_B = (size_t)256 * 64 * 128 * 2;   // 4 MB (f16 window buf)
    const size_t TOK_H = (size_t)12544 * 128 * 2;      // f16 token buf
    const size_t TOK_F = (size_t)12544 * 128 * 4;      // f32 token buf

    _Float16* xw  = (_Float16*)(ws + alloc(WIN_B));    // later reused as a2
    _Float16* qw  = (_Float16*)(ws + alloc(WIN_B));    // later q2
    _Float16* kw  = (_Float16*)(ws + alloc(WIN_B));    // later k2
    _Float16* vw  = (_Float16*)(ws + alloc(WIN_B));    // later v2
    _Float16* aw  = (_Float16*)(ws + alloc(WIN_B));    // later ln2
    float*    x1  = (float*)(ws + alloc(TOK_F));
    _Float16* ln1 = (_Float16*)(ws + alloc(TOK_H));
    float*    x2  = (float*)(ws + alloc(TOK_F));
    _Float16* hb  = (_Float16*)(ws + alloc((size_t)12544 * 512 * 2));
    _Float16* wqT = (_Float16*)(ws + alloc((size_t)128 * 128 * 2));
    _Float16* wkT = (_Float16*)(ws + alloc((size_t)128 * 128 * 2));
    _Float16* wvT = (_Float16*)(ws + alloc((size_t)128 * 128 * 2));
    _Float16* woT = (_Float16*)(ws + alloc((size_t)128 * 128 * 2));
    _Float16* w1T = (_Float16*)(ws + alloc((size_t)512 * 128 * 2));
    _Float16* w2T = (_Float16*)(ws + alloc((size_t)128 * 512 * 2));
    _Float16* q2 = qw; _Float16* k2 = kw; _Float16* v2 = vw;
    _Float16* a2 = xw; _Float16* ln2 = aw;

    dim3 blk(256);

    // 0) weights -> f16, transposed [N][K]
    wtrans_kernel<<<dim3(64),  blk, 0, stream>>>(wq, wqT, 128, 128);
    wtrans_kernel<<<dim3(64),  blk, 0, stream>>>(wk, wkT, 128, 128);
    wtrans_kernel<<<dim3(64),  blk, 0, stream>>>(wv, wvT, 128, 128);
    wtrans_kernel<<<dim3(64),  blk, 0, stream>>>(wo, woT, 128, 128);
    wtrans_kernel<<<dim3(256), blk, 0, stream>>>(w1, w1T, 128, 512);
    wtrans_kernel<<<dim3(256), blk, 0, stream>>>(w2, w2T, 512, 128);

    // 1) shift + window partition + f16 convert
    shift_gather_kernel<<<dim3(256), blk, 0, stream>>>(x, xw);

    // 2) windowed Q,K,V projections  (M=16384, N=128, K=128)
    gemm_kernel<<<dim3(2, 128), blk, 0, stream>>>(xw, wqT, bq, nullptr, qw, 16384, 128, 128, F_F16);
    gemm_kernel<<<dim3(2, 128), blk, 0, stream>>>(xw, wkT, bk, nullptr, kw, 16384, 128, 128, F_F16);
    gemm_kernel<<<dim3(2, 128), blk, 0, stream>>>(xw, wvT, bv, nullptr, vw, 16384, 128, 128, F_F16);

    // 3) window attention (256 windows x 2 head-pairs)
    win_attn_kernel<<<dim3(512), blk, 0, stream>>>(qw, kw, vw, aw);

    // 4) output projection + window reverse + un-shift scatter -> x1 (f32)
    gemm_kernel<<<dim3(2, 128), blk, 0, stream>>>(aw, woT, bo, nullptr, x1, 16384, 128, 128, F_SCAT);

    // 5) LN1
    layernorm_kernel<<<dim3(1568), blk, 0, stream>>>(x1, g1, be1, ln1);

    // 6) global Q,K,V projections  (M=12544)
    gemm_kernel<<<dim3(2, 98), blk, 0, stream>>>(ln1, wqT, bq, nullptr, q2, 12544, 128, 128, F_F16);
    gemm_kernel<<<dim3(2, 98), blk, 0, stream>>>(ln1, wkT, bk, nullptr, k2, 12544, 128, 128, F_F16);
    gemm_kernel<<<dim3(2, 98), blk, 0, stream>>>(ln1, wvT, bv, nullptr, v2, 12544, 128, 128, F_F16);

    // 7) global attention, flash-style (4 batches x 4 heads x 49 q-blocks)
    global_attn_kernel<<<dim3(784), blk, 0, stream>>>(q2, k2, v2, a2);

    // 8) output projection + residual(x1) -> x2 (f32)
    gemm_kernel<<<dim3(2, 98), blk, 0, stream>>>(a2, woT, bo, x1, x2, 12544, 128, 128, F_RES);

    // 9) LN2
    layernorm_kernel<<<dim3(1568), blk, 0, stream>>>(x2, g2, be2, ln2);

    // 10) MLP fc1 + GELU -> hb (f16)  (N=512)
    gemm_kernel<<<dim3(8, 98), blk, 0, stream>>>(ln2, w1T, b1, nullptr, hb, 12544, 512, 128, F_F16 | F_GELU);

    // 11) MLP fc2 + residual(x2) -> d_out (f32)  (K=512)
    gemm_kernel<<<dim3(2, 98), blk, 0, stream>>>(hb, w2T, b2, x2, d_out, 12544, 128, 512, F_RES);
}
